// SNNForecasterV6_83485574300332
// MI455X (gfx1250) — compile-verified, matrix operand order checked by reading, not run on hardware
//
#include <hip/hip_runtime.h>
#include <hip/hip_bf16.h>
#include <math.h>

// ---------------------------------------------------------------------------
// SNN forecaster pipeline for MI455X (gfx1250, wave32, WMMA).
//  conv1d -> BN -> [LIF -> perstep-BN -> LIF -> perstep-BN -> LIF] (fused)
//  -> Q/K/V bf16 WMMA GEMMs (32x32 register-blocked per wave)
//  -> fused sigmoid(QK^T)*colsum (WMMA, B-fragment reuse x4)
//  -> feat = (colsum/T) @ V -> batch BN -> head linear.
// ---------------------------------------------------------------------------

typedef __attribute__((ext_vector_type(16))) __bf16 v16bf;
typedef __attribute__((ext_vector_type(8)))  float  v8f;

union FragBF { v16bf v; uint4 q[2]; };

constexpr int Bn   = 128;        // batch
constexpr int Tn   = 128;        // raw timesteps
constexpr int Cn   = 32;         // input channels
constexpr int Hn   = 512;        // hidden
constexpr int Tsn  = 4;          // time expansion
constexpr int HTn  = Hn * Tsn;   // 2048 conv channels
constexpr int TT   = Tn * Tsn;   // 512 total scan steps
constexpr int NROW = TT * Bn;    // 65536 rows for attention GEMMs
constexpr int HOR  = 96;
constexpr float EPSf = 1e-5f;

__device__ __forceinline__ unsigned short f2bf(float f) {
  unsigned u = __float_as_uint(f);
  u += 0x7FFFu + ((u >> 16) & 1u);          // round-to-nearest-even
  return (unsigned short)(u >> 16);
}
__device__ __forceinline__ float bf2f(unsigned short h) {
  return __uint_as_float(((unsigned)h) << 16);
}
__device__ __forceinline__ float sigmoidf(float x) {
  return 1.f / (1.f + __expf(-x));
}

// ---------------------------------------------------------------------------
// Kernel 1: Conv1d(C->HT, k=3, pad=1) + bias, write [o][t][b] layout,
// plus deterministic block-partial sums for BN1 stats.
// grid (HTn, 64), block (128, 2)
// ---------------------------------------------------------------------------
__global__ void k_conv(const float* __restrict__ x, const float* __restrict__ w,
                       const float* __restrict__ cb, float* __restrict__ out,
                       float* __restrict__ psum, float* __restrict__ psq) {
  __shared__ float s_w[Cn * 3];
  __shared__ float s_a[256];
  __shared__ float s_b[256];
  const int o   = blockIdx.x;
  const int b   = threadIdx.x;                    // 0..127
  const int t   = blockIdx.y * 2 + threadIdx.y;   // 0..127
  const int tid = threadIdx.y * 128 + threadIdx.x;
  if (tid < Cn * 3) s_w[tid] = w[o * (Cn * 3) + tid];
  __syncthreads();
  float acc = cb[o];
#pragma unroll
  for (int kk = 0; kk < 3; ++kk) {
    const int tt = t + kk - 1;
    if (tt >= 0 && tt < Tn) {
      const float* xr = x + ((size_t)b * Tn + tt) * Cn;
#pragma unroll
      for (int c = 0; c < Cn; ++c) acc = fmaf(xr[c], s_w[c * 3 + kk], acc);
    }
  }
  out[((size_t)o * Tn + t) * Bn + b] = acc;
  s_a[tid] = acc;
  s_b[tid] = acc * acc;
  __syncthreads();
  for (int s = 128; s > 0; s >>= 1) {
    if (tid < s) { s_a[tid] += s_a[tid + s]; s_b[tid] += s_b[tid + s]; }
    __syncthreads();
  }
  if (tid == 0) {
    psum[o * 64 + blockIdx.y] = s_a[0];
    psq [o * 64 + blockIdx.y] = s_b[0];
  }
}

// ---------------------------------------------------------------------------
// Kernel 2: finalize BN1 -> per-channel scale/shift (deterministic order)
// ---------------------------------------------------------------------------
__global__ void k_bn1(const float* __restrict__ psum, const float* __restrict__ psq,
                      const float* __restrict__ g, const float* __restrict__ bb,
                      float* __restrict__ sc, float* __restrict__ sh) {
  const int o = blockIdx.x * blockDim.x + threadIdx.x;
  if (o >= HTn) return;
  float s = 0.f, q = 0.f;
  for (int i = 0; i < 64; ++i) { s += psum[o * 64 + i]; q += psq[o * 64 + i]; }
  const float inv_n = 1.0f / (float)(Bn * Tn);
  const float m = s * inv_n;
  const float v = q * inv_n - m * m;
  const float scale = g[o] * rsqrtf(v + EPSf);
  sc[o] = scale;
  sh[o] = bb[o] - m * scale;
}

// ---------------------------------------------------------------------------
// Kernel 3: fused triple LIF scan + two per-step BNs over spikes.
// Spikes are {0,1} so batch BN reduces to m = popcount/128, v = m - m^2.
// block = 128 batch x 2 features (8 waves); grid = Hn/2 = 256 blocks.
// Output mem3 as bf16 in flat-row layout m = step*B + b.
// ---------------------------------------------------------------------------
__global__ void k_scan(const float* __restrict__ conv, const float* __restrict__ sc1,
                       const float* __restrict__ sh1, const float* __restrict__ beta_e,
                       const float* __restrict__ g2, const float* __restrict__ b2,
                       const float* __restrict__ beta2, const float* __restrict__ g3,
                       const float* __restrict__ b3, const float* __restrict__ beta3,
                       unsigned short* __restrict__ xa) {
  __shared__ float s_c1[8];
  __shared__ float s_c2[8];
  const int tx   = threadIdx.x;
  const int b    = tx & 127;
  const int hl   = tx >> 7;     // 0..1
  const int wv   = tx >> 5;     // wave 0..7
  const int lane = tx & 31;
  const int h    = blockIdx.x * 2 + hl;
  float c1s[Tsn], c1h[Tsn];
#pragma unroll
  for (int ts = 0; ts < Tsn; ++ts) {
    const int o = ts * Hn + h;
    c1s[ts] = sc1[o];
    c1h[ts] = sh1[o];
  }
  const float be  = fminf(fmaxf(beta_e[h], 0.f), 0.99f);
  const float b2c = fminf(fmaxf(beta2[h], 0.f), 0.99f);
  const float b3c = fminf(fmaxf(beta3[h], 0.f), 0.99f);
  const float g2h = g2[h], b2h = b2[h], g3h = g3[h], b3h = b3[h];
  float m1 = 0.f, m2 = 0.f, m3 = 0.f, p1 = 0.f, p2 = 0.f, p3 = 0.f;
  for (int step = 0; step < TT; ++step) {
    const int t = step >> 2, ts = step & 3;
    float cur = conv[(((size_t)(ts * Hn + h)) * Tn + t) * Bn + b];
    cur = fmaf(cur, c1s[ts], c1h[ts]);
    // LIF1 (reset == previous spike; subtract-reset, THR=1)
    m1 = be * m1 + cur - p1;
    const float s1 = (m1 > 1.f) ? 1.f : 0.f;
    p1 = s1;
    unsigned bal = __builtin_amdgcn_ballot_w32(m1 > 1.f);
    if (lane == 0) s_c1[wv] = (float)__builtin_popcount(bal);
    __syncthreads();
    float cnt = s_c1[hl * 4] + s_c1[hl * 4 + 1] + s_c1[hl * 4 + 2] + s_c1[hl * 4 + 3];
    float mu  = cnt * (1.f / 128.f);
    float var = mu - mu * mu;                 // Bernoulli: E[s^2]=E[s]
    const float h2 = (s1 - mu) * rsqrtf(var + EPSf) * g2h + b2h;
    // LIF2
    m2 = b2c * m2 + h2 - p2;
    const float s2 = (m2 > 1.f) ? 1.f : 0.f;
    p2 = s2;
    bal = __builtin_amdgcn_ballot_w32(m2 > 1.f);
    if (lane == 0) s_c2[wv] = (float)__builtin_popcount(bal);
    __syncthreads();
    cnt = s_c2[hl * 4] + s_c2[hl * 4 + 1] + s_c2[hl * 4 + 2] + s_c2[hl * 4 + 3];
    mu  = cnt * (1.f / 128.f);
    var = mu - mu * mu;
    const float h3 = (s2 - mu) * rsqrtf(var + EPSf) * g3h + b3h;
    // LIF3 (keep membrane trajectory)
    m3 = b3c * m3 + h3 - p3;
    p3 = (m3 > 1.f) ? 1.f : 0.f;
    xa[((size_t)step * Bn + b) * Hn + h] = f2bf(m3);
  }
}

// ---------------------------------------------------------------------------
// Kernel 4: fp32 -> bf16 convert (for weight matrices)
// ---------------------------------------------------------------------------
__global__ void k_f2bf(const float* __restrict__ src, unsigned short* __restrict__ dst, int n) {
  const int i = blockIdx.x * blockDim.x + threadIdx.x;
  if (i < n) dst[i] = f2bf(src[i]);
}

// ---------------------------------------------------------------------------
// Kernel 5: Y = X @ W^T + bias, bf16 in / bf16 out, fp32 accumulate via
// v_wmma_f32_16x16x32_bf16.  M = NROW, N = K = 512.
// Register-blocked: one 32x32 macro-tile (2x2 accumulators) per wave.
// Per k-step: 2 A-frags + 2 B-frags -> 4 WMMAs (A and B each reused 2x).
// A fragment per-lane layout (16-bit, 16x32): lane L holds row M=L%16,
// K(j) = (j&7) + (j>=8?16:0) + (L>=16?8:0)  -> two contiguous 16B chunks.
// B operand holds W rows (column-major B == row-major W), same pattern.
// ---------------------------------------------------------------------------
__global__ void k_gemm(const unsigned short* __restrict__ A,
                       const unsigned short* __restrict__ W,
                       const float* __restrict__ bias,
                       unsigned short* __restrict__ Y) {
  const int lane = threadIdx.x & 31;
  const int wv   = threadIdx.x >> 5;
  const int tile = blockIdx.x * 8 + wv;            // 32x32 macro-tile id
  const int nmac = tile & 15, mmac = tile >> 4;    // Hn/32 = 16 n-macro-tiles
  const int m0 = mmac * 32, n0 = nmac * 32;
  const int r15  = lane & 15;
  const int koff = (lane >= 16) ? 8 : 0;
  const unsigned short* pa0 = A + ((size_t)(m0 + r15)) * Hn + koff;
  const unsigned short* pa1 = pa0 + (size_t)16 * Hn;
  const unsigned short* pb0 = W + ((size_t)(n0 + r15)) * Hn + koff;
  const unsigned short* pb1 = pb0 + (size_t)16 * Hn;
  __builtin_prefetch(pa0 + 256, 0, 3);
  __builtin_prefetch(pa1 + 256, 0, 3);
  v8f a00 = {}, a01 = {}, a10 = {}, a11 = {};
#pragma unroll
  for (int kb = 0; kb < 16; ++kb) {
    const int k0 = kb * 32;
    FragBF fa0, fa1, fb0, fb1;
    fa0.q[0] = *(const uint4*)(pa0 + k0); fa0.q[1] = *(const uint4*)(pa0 + k0 + 16);
    fa1.q[0] = *(const uint4*)(pa1 + k0); fa1.q[1] = *(const uint4*)(pa1 + k0 + 16);
    fb0.q[0] = *(const uint4*)(pb0 + k0); fb0.q[1] = *(const uint4*)(pb0 + k0 + 16);
    fb1.q[0] = *(const uint4*)(pb1 + k0); fb1.q[1] = *(const uint4*)(pb1 + k0 + 16);
    a00 = __builtin_amdgcn_wmma_f32_16x16x32_bf16(false, fa0.v, false, fb0.v,
                                                  (short)0, a00, false, false);
    a01 = __builtin_amdgcn_wmma_f32_16x16x32_bf16(false, fa0.v, false, fb1.v,
                                                  (short)0, a01, false, false);
    a10 = __builtin_amdgcn_wmma_f32_16x16x32_bf16(false, fa1.v, false, fb0.v,
                                                  (short)0, a10, false, false);
    a11 = __builtin_amdgcn_wmma_f32_16x16x32_bf16(false, fa1.v, false, fb1.v,
                                                  (short)0, a11, false, false);
  }
  const int   mb  = m0 + ((lane >= 16) ? 8 : 0);
  const float bn0 = bias[n0 + r15];
  const float bn1 = bias[n0 + 16 + r15];
#pragma unroll
  for (int r = 0; r < 8; ++r) {
    Y[((size_t)(mb + r))      * Hn + n0 + r15]      = f2bf(a00[r] + bn0);
    Y[((size_t)(mb + r))      * Hn + n0 + 16 + r15] = f2bf(a01[r] + bn1);
    Y[((size_t)(mb + 16 + r)) * Hn + n0 + r15]      = f2bf(a10[r] + bn0);
    Y[((size_t)(mb + 16 + r)) * Hn + n0 + 16 + r15] = f2bf(a11[r] + bn1);
  }
}

// ---------------------------------------------------------------------------
// Kernel 6: per-batch  acol[b][s] = sum_t sigmoid(Q_b K_b^T * H^-0.5)[t][s]
// (column sums of the attention matrix; attn itself is never stored).
// grid = B*32 (batch, n-tile); wave owns m-tiles {wv, wv+8, wv+16, wv+24}
// with 4 accumulators so the K(B) fragment is loaded ONCE per k-step.
// ---------------------------------------------------------------------------
__global__ void k_attn_colsum(const unsigned short* __restrict__ Q,
                              const unsigned short* __restrict__ Kt,
                              float* __restrict__ acol) {
  __shared__ float s_cs[8][16];
  const int lane = threadIdx.x & 31;
  const int wv   = threadIdx.x >> 5;
  const int bb   = blockIdx.x >> 5;
  const int n0   = (blockIdx.x & 31) * 16;
  const int r15  = lane & 15;
  const int koff = (lane >= 16) ? 8 : 0;
  const unsigned short* pb = Kt + (((size_t)(n0 + r15) * Bn + bb) * Hn) + koff;
  // Q rows for the wave's four m-tiles (row t lives at flat row t*Bn + bb)
  const unsigned short* pa0 = Q + (((size_t)((wv +  0) * 16 + r15) * Bn + bb) * Hn) + koff;
  const unsigned short* pa1 = Q + (((size_t)((wv +  8) * 16 + r15) * Bn + bb) * Hn) + koff;
  const unsigned short* pa2 = Q + (((size_t)((wv + 16) * 16 + r15) * Bn + bb) * Hn) + koff;
  const unsigned short* pa3 = Q + (((size_t)((wv + 24) * 16 + r15) * Bn + bb) * Hn) + koff;
  v8f ac0 = {}, ac1 = {}, ac2 = {}, ac3 = {};
#pragma unroll
  for (int kb = 0; kb < 16; ++kb) {
    const int k0 = kb * 32;
    FragBF fb, f0, f1, f2, f3;
    fb.q[0] = *(const uint4*)(pb  + k0); fb.q[1] = *(const uint4*)(pb  + k0 + 16);
    f0.q[0] = *(const uint4*)(pa0 + k0); f0.q[1] = *(const uint4*)(pa0 + k0 + 16);
    f1.q[0] = *(const uint4*)(pa1 + k0); f1.q[1] = *(const uint4*)(pa1 + k0 + 16);
    f2.q[0] = *(const uint4*)(pa2 + k0); f2.q[1] = *(const uint4*)(pa2 + k0 + 16);
    f3.q[0] = *(const uint4*)(pa3 + k0); f3.q[1] = *(const uint4*)(pa3 + k0 + 16);
    ac0 = __builtin_amdgcn_wmma_f32_16x16x32_bf16(false, f0.v, false, fb.v,
                                                  (short)0, ac0, false, false);
    ac1 = __builtin_amdgcn_wmma_f32_16x16x32_bf16(false, f1.v, false, fb.v,
                                                  (short)0, ac1, false, false);
    ac2 = __builtin_amdgcn_wmma_f32_16x16x32_bf16(false, f2.v, false, fb.v,
                                                  (short)0, ac2, false, false);
    ac3 = __builtin_amdgcn_wmma_f32_16x16x32_bf16(false, f3.v, false, fb.v,
                                                  (short)0, ac3, false, false);
  }
  const float sc = 0.04419417382415922f;  // 512^-0.5
  float csum = 0.f;
#pragma unroll
  for (int r = 0; r < 8; ++r) {
    csum += sigmoidf(ac0[r] * sc);
    csum += sigmoidf(ac1[r] * sc);
    csum += sigmoidf(ac2[r] * sc);
    csum += sigmoidf(ac3[r] * sc);
  }
  csum += __shfl_xor(csum, 16, 32);         // fold row-halves (lanes L, L+16)
  if (lane < 16) s_cs[wv][lane] = csum;
  __syncthreads();
  if (threadIdx.x < 16) {
    float tot = 0.f;
    for (int w = 0; w < 8; ++w) tot += s_cs[w][threadIdx.x];
    acol[(size_t)bb * Hn + n0 + threadIdx.x] = tot;
  }
}

// ---------------------------------------------------------------------------
// Kernel 7: feat[b][h] = (1/T) * sum_s acol[b][s] * V[b][s][h]
// grid = B; 256 threads cover 2 h each; V reads coalesced.
// ---------------------------------------------------------------------------
__global__ void k_feat(const float* __restrict__ acol,
                       const unsigned short* __restrict__ V,
                       float* __restrict__ feat) {
  __shared__ float s_a[TT];
  const int bb = blockIdx.x;
  for (int i = threadIdx.x; i < TT; i += 256)
    s_a[i] = acol[(size_t)bb * Hn + i] * (1.f / (float)TT);
  __syncthreads();
  const int h0 = threadIdx.x, h1 = threadIdx.x + 256;
  float a0 = 0.f, a1 = 0.f;
  for (int s = 0; s < TT; ++s) {
    const unsigned short* vr = V + (((size_t)s * Bn) + bb) * Hn;
    const float w = s_a[s];
    a0 = fmaf(w, bf2f(vr[h0]), a0);
    a1 = fmaf(w, bf2f(vr[h1]), a1);
  }
  feat[(size_t)bb * Hn + h0] = a0;
  feat[(size_t)bb * Hn + h1] = a1;
}

// ---------------------------------------------------------------------------
// Kernel 8: batch-norm stats over feat (per h) -> scale/shift
// ---------------------------------------------------------------------------
__global__ void k_featbn(const float* __restrict__ feat, const float* __restrict__ g,
                         const float* __restrict__ bb_, float* __restrict__ fsc,
                         float* __restrict__ fsh) {
  const int h = blockIdx.x * blockDim.x + threadIdx.x;
  if (h >= Hn) return;
  float s = 0.f, q = 0.f;
  for (int b = 0; b < Bn; ++b) {
    const float v = feat[(size_t)b * Hn + h];
    s += v; q += v * v;
  }
  const float m   = s * (1.f / (float)Bn);
  const float var = q * (1.f / (float)Bn) - m * m;
  const float scl = g[h] * rsqrtf(var + EPSf);
  fsc[h] = scl;
  fsh[h] = bb_[h] - m * scl;
}

// ---------------------------------------------------------------------------
// Kernel 9: head  out[b][j] = sum_h featn[b][h] * wh[j][h] + bh[j]
// grid = B; 96 threads.
// ---------------------------------------------------------------------------
__global__ void k_head(const float* __restrict__ feat, const float* __restrict__ fsc,
                       const float* __restrict__ fsh, const float* __restrict__ wh,
                       const float* __restrict__ bh, float* __restrict__ out) {
  __shared__ float s_f[Hn];
  const int bb = blockIdx.x;
  for (int i = threadIdx.x; i < Hn; i += blockDim.x)
    s_f[i] = fmaf(feat[(size_t)bb * Hn + i], fsc[i], fsh[i]);
  __syncthreads();
  const int j = threadIdx.x;
  float acc = bh[j];
  for (int hh = 0; hh < Hn; ++hh)
    acc = fmaf(s_f[hh], wh[(size_t)j * Hn + hh], acc);
  out[(size_t)bb * HOR + j] = acc;
}

// ---------------------------------------------------------------------------
// Host launcher
// ---------------------------------------------------------------------------
extern "C" void kernel_launch(void* const* d_in, const int* in_sizes, int n_in,
                              void* d_out, int out_size, void* d_ws, size_t ws_size,
                              hipStream_t stream) {
  const float* x        = (const float*)d_in[0];
  const float* conv_w   = (const float*)d_in[1];
  const float* conv_b   = (const float*)d_in[2];
  const float* bn1_g    = (const float*)d_in[3];
  const float* bn1_b    = (const float*)d_in[4];
  const float* beta_enc = (const float*)d_in[5];
  const float* bn2_g    = (const float*)d_in[6];
  const float* bn2_b    = (const float*)d_in[7];
  const float* beta2    = (const float*)d_in[8];
  const float* bn3_g    = (const float*)d_in[9];
  const float* bn3_b    = (const float*)d_in[10];
  const float* beta3    = (const float*)d_in[11];
  const float* wq       = (const float*)d_in[12];
  const float* bq       = (const float*)d_in[13];
  const float* wk       = (const float*)d_in[14];
  const float* bk       = (const float*)d_in[15];
  const float* wv       = (const float*)d_in[16];
  const float* bv       = (const float*)d_in[17];
  const float* bna_g    = (const float*)d_in[18];
  const float* bna_b    = (const float*)d_in[19];
  const float* wh       = (const float*)d_in[20];
  const float* bh       = (const float*)d_in[21];
  float* out = (float*)d_out;

  // Workspace carve-out (256B aligned slabs).
  char* p = (char*)d_ws;
  auto carve = [&](size_t bytes) -> void* {
    void* r = (void*)p;
    p += (bytes + 255) & ~(size_t)255;
    return r;
  };
  float* convb = (float*)carve(sizeof(float) * (size_t)HTn * Tn * Bn);   // 128 MB
  float* psum  = (float*)carve(sizeof(float) * (size_t)HTn * 64);
  float* psq   = (float*)carve(sizeof(float) * (size_t)HTn * 64);
  float* sc1   = (float*)carve(sizeof(float) * HTn);
  float* sh1   = (float*)carve(sizeof(float) * HTn);
  float* fsc   = (float*)carve(sizeof(float) * Hn);
  float* fsh   = (float*)carve(sizeof(float) * Hn);
  float* acol  = (float*)carve(sizeof(float) * (size_t)Bn * Hn);
  float* feat  = (float*)carve(sizeof(float) * (size_t)Bn * Hn);
  unsigned short* xa  = (unsigned short*)carve(2ull * NROW * Hn);        // 64 MB
  unsigned short* qb  = (unsigned short*)carve(2ull * NROW * Hn);
  unsigned short* kb_ = (unsigned short*)carve(2ull * NROW * Hn);
  unsigned short* vb  = (unsigned short*)carve(2ull * NROW * Hn);
  unsigned short* wqb = (unsigned short*)carve(2ull * Hn * Hn);
  unsigned short* wkb = (unsigned short*)carve(2ull * Hn * Hn);
  unsigned short* wvb = (unsigned short*)carve(2ull * Hn * Hn);

  // 1) Conv + bias + BN1 partial stats
  k_conv<<<dim3(HTn, 64), dim3(128, 2), 0, stream>>>(x, conv_w, conv_b, convb, psum, psq);
  // 2) BN1 finalize -> per-channel affine
  k_bn1<<<8, 256, 0, stream>>>(psum, psq, bn1_g, bn1_b, sc1, sh1);
  // 3) Fused triple LIF scan + per-step BNs -> mem3 (bf16)
  k_scan<<<Hn / 2, 256, 0, stream>>>(convb, sc1, sh1, beta_enc, bn2_g, bn2_b, beta2,
                                     bn3_g, bn3_b, beta3, xa);
  // 4) Weight conversion to bf16
  k_f2bf<<<(Hn * Hn) / 256, 256, 0, stream>>>(wq, wqb, Hn * Hn);
  k_f2bf<<<(Hn * Hn) / 256, 256, 0, stream>>>(wk, wkb, Hn * Hn);
  k_f2bf<<<(Hn * Hn) / 256, 256, 0, stream>>>(wv, wvb, Hn * Hn);
  // 5) Q/K/V WMMA GEMMs: (65536 x 512) @ (512 x 512)^T, 32x32 per wave
  const int gemm_blocks = (NROW / 32) * (Hn / 32) / 8;  // 4096
  k_gemm<<<gemm_blocks, 256, 0, stream>>>(xa, wqb, bq, qb);
  k_gemm<<<gemm_blocks, 256, 0, stream>>>(xa, wkb, bk, kb_);
  k_gemm<<<gemm_blocks, 256, 0, stream>>>(xa, wvb, bv, vb);
  // 6) Fused sigmoid(QK^T)*colsum per batch (WMMA)
  k_attn_colsum<<<Bn * 32, 256, 0, stream>>>(qb, kb_, acol);
  // 7) feat = (colsum/T) @ V
  k_feat<<<Bn, 256, 0, stream>>>(acol, vb, feat);
  // 8) Batch-norm over features
  k_featbn<<<2, 256, 0, stream>>>(feat, bna_g, bna_b, fsc, fsh);
  // 9) Head linear
  k_head<<<Bn, 96, 0, stream>>>(feat, fsc, fsh, wh, bh, out);

  (void)in_sizes; (void)n_in; (void)out_size; (void)ws_size;
}
